// Dense_7370163880186
// MI455X (gfx1250) — matrix operand order repaired and checked by reading, not compiled, not run on hardware
//
#include <hip/hip_runtime.h>
#include <cmath>

// ---- problem constants (match reference) ----
#define B_    64
#define L_    8192
#define CIN_  128
#define COUT_ 128

// padded LDS row strides: 136 halves = 68 dwords, 68 % 64 == 4 -> conflict-free
// b128 reads when 16 lanes read 16 consecutive rows.
#define XS_STR 136   // halves per x row in LDS (128 + 8)
#define WS_STR 136   // halves per w row in LDS (128 + 8)

typedef __attribute__((ext_vector_type(16))) _Float16 v16h;
typedef __attribute__((ext_vector_type(8)))  _Float16 v8h;
typedef __attribute__((ext_vector_type(4)))  _Float16 v4h;
typedef __attribute__((ext_vector_type(8)))  float    v8f;
typedef __attribute__((ext_vector_type(4)))  float    v4f;

// One workgroup per site l. 256 threads = 8 wave32.
// Per wave: 32(M) x 32(N) output tile, K = 128 in 4 steps of 32.
__global__ __launch_bounds__(256) void site_gemm_kernel(
    const float* __restrict__ x,            // (B, L, CIN) f32
    const long long* __restrict__ channels, // (L,) i64
    const float* __restrict__ weight,       // (NCH, COUT, CIN) f32
    const float* __restrict__ bias,         // (NCH, COUT) f32
    float* __restrict__ out)                // (B, L, COUT) f32
{
    __shared__ _Float16 xs[B_ * XS_STR];      // x tile, f16 (fragments)
    __shared__ _Float16 ws[COUT_ * WS_STR];   // weight tile, f16
    __shared__ float    bs[COUT_];

    const int l   = blockIdx.x;
    const int tid = threadIdx.x;
    const long long ch = channels[l];

    // ---- stage W[ch] (128x128 f32 -> f16 LDS), 16384 floats, 64/thread ----
    {
        const float* wsrc = weight + (size_t)ch * (COUT_ * CIN_);
        #pragma unroll
        for (int i = 0; i < 16; ++i) {
            int f4 = tid + i * 256;           // float4 unit index, 0..4095
            int o  = f4 >> 5;                 // 32 float4 per row
            int c  = (f4 & 31) << 2;
            v4f w4 = *(const v4f*)(wsrc + (size_t)o * CIN_ + c);
            v4h h4;
            h4.x = (_Float16)w4.x; h4.y = (_Float16)w4.y;
            h4.z = (_Float16)w4.z; h4.w = (_Float16)w4.w;
            *(v4h*)(&ws[o * WS_STR + c]) = h4;
        }
    }

    // ---- stage x[:, l, :] (64x128 f32 -> f16 LDS), 8192 floats, 32/thread ----
    {
        #pragma unroll
        for (int i = 0; i < 8; ++i) {
            int f4 = tid + i * 256;           // float4 unit index, 0..2047
            int b  = f4 >> 5;
            int c  = (f4 & 31) << 2;
            v4f xv = *(const v4f*)(x + (size_t)b * (L_ * CIN_) + (size_t)l * CIN_ + c);
            v4h h4;
            h4.x = (_Float16)xv.x; h4.y = (_Float16)xv.y;
            h4.z = (_Float16)xv.z; h4.w = (_Float16)xv.w;
            *(v4h*)(&xs[b * XS_STR + c]) = h4;
        }
    }

    if (tid < COUT_) bs[tid] = bias[(size_t)ch * COUT_ + tid];

    __syncthreads();

    const int wave = tid >> 5;
    const int lane = tid & 31;
    const int lr   = lane & 15;   // 0..15
    const int lh   = lane >> 4;   // 0/1
    const int mo   = (wave >> 2) * 32;   // M offset (0 or 32)
    const int no   = (wave & 3) * 32;    // N offset (0,32,64,96)

    v8f acc[2][2] = {};

    #pragma unroll
    for (int kt = 0; kt < 4; ++kt) {
        const int k0 = kt * 32;

        // A fragments: 16x32 f16, A[m][k] = x[b=m][c=k]
        // lane<16: halves 0..7 = K k0..k0+7, halves 8..15 = K k0+16..k0+23
        // lane>=16: +8 on K within each group  -> two contiguous b128 LDS reads
        v16h a[2];
        #pragma unroll
        for (int mt = 0; mt < 2; ++mt) {
            const _Float16* row = &xs[(mo + mt * 16 + lr) * XS_STR + k0 + lh * 8];
            v8h p0 = *(const v8h*)(row);        // K = k0 + lh*8 + 0..7
            v8h p1 = *(const v8h*)(row + 16);   // K = k0 + 16 + lh*8 + 0..7
            a[mt] = __builtin_shufflevector(p0, p1,
                0, 1, 2, 3, 4, 5, 6, 7, 8, 9, 10, 11, 12, 13, 14, 15);
        }

        // B fragments: 32x16 f16, B[k][n] = W[o=n][c=k] (W row contiguous in k)
        // lane: n = lr; halves 0..7 -> K = k0 + lh*8 + i, halves 8..15 -> K = k0 + 16 + lh*8 + (i-8)
        v16h bfrag[2];
        #pragma unroll
        for (int nt = 0; nt < 2; ++nt) {
            const _Float16* wr = &ws[(no + nt * 16 + lr) * WS_STR + k0 + lh * 8];
            v8h p0 = *(const v8h*)(wr);
            v8h p1 = *(const v8h*)(wr + 16);
            bfrag[nt] = __builtin_shufflevector(p0, p1,
                0, 1, 2, 3, 4, 5, 6, 7, 8, 9, 10, 11, 12, 13, 14, 15);
        }

        #pragma unroll
        for (int mt = 0; mt < 2; ++mt)
            #pragma unroll
            for (int nt = 0; nt < 2; ++nt)
                acc[mt][nt] = __builtin_amdgcn_wmma_f32_16x16x32_f16(
                    /*neg_a=*/false, a[mt], /*neg_b=*/false, bfrag[nt],
                    /*c_mod=*/(short)0, acc[mt][nt],
                    /*reuse_a=*/false, /*reuse_b=*/false);
    }

    // ---- epilogue: tanh(acc + bias) + x, write (B, L, COUT) ----
    // Residual x re-read from global in exact f32; the 32KB tile was read
    // moments ago by this block, so it is L2-resident (192 MB L2).
    #pragma unroll
    for (int mt = 0; mt < 2; ++mt) {
        #pragma unroll
        for (int nt = 0; nt < 2; ++nt) {
            const int n  = no + nt * 16 + lr;
            const float bv = bs[n];
            #pragma unroll
            for (int i = 0; i < 8; ++i) {
                const int m = mo + mt * 16 + lh * 8 + i;  // C layout: vgpr i, half lh
                const size_t idx = (size_t)m * (L_ * CIN_) + (size_t)l * CIN_ + n;
                float v = tanhf(acc[mt][nt][i] + bv) + x[idx];
                out[idx] = v;
            }
        }
    }
}

// Second tuple output: channels, appended after y in d_out.
// mode 0: raw int64 bytes (tail == 2*L floats); mode 1: cast to float (tail == L).
__global__ void copy_channels_kernel(const long long* __restrict__ ch,
                                     float* __restrict__ tail, int mode)
{
    int i = blockIdx.x * 256 + threadIdx.x;
    if (i < L_) {
        if (mode == 0) ((long long*)tail)[i] = ch[i];
        else           tail[i] = (float)ch[i];
    }
}

extern "C" void kernel_launch(void* const* d_in, const int* in_sizes, int n_in,
                              void* d_out, int out_size, void* d_ws, size_t ws_size,
                              hipStream_t stream)
{
    const float*     x        = (const float*)d_in[0];
    const long long* channels = (const long long*)d_in[1];
    const float*     weight   = (const float*)d_in[2];
    const float*     bias     = (const float*)d_in[3];
    float*           out      = (float*)d_out;

    site_gemm_kernel<<<L_, 256, 0, stream>>>(x, channels, weight, bias, out);

    const long long y_elems = (long long)B_ * L_ * COUT_;
    const long long tail = (long long)out_size - y_elems;
    if (tail >= 2 * (long long)L_) {
        copy_channels_kernel<<<(L_ + 255) / 256, 256, 0, stream>>>(
            channels, out + (size_t)y_elems, 0);
    } else if (tail >= (long long)L_) {
        copy_channels_kernel<<<(L_ + 255) / 256, 256, 0, stream>>>(
            channels, out + (size_t)y_elems, 1);
    }
}
